// ViT_87497073754963
// MI455X (gfx1250) — compile-verified, matrix operand order checked by reading, not run on hardware
//
#include <hip/hip_runtime.h>
#include <hip/hip_bf16.h>
#include <math.h>

// ---- ViT-Base constants (match reference) ----
#define BVIT   16
#define CCH    3
#define IMGSZ  224
#define NPAT   14
#define PSZ    16
#define DIM    768
#define NHEAD  12
#define LAYERS 12
#define OUTC   1000
#define DHEAD  64
#define MLPD   3072
#define SEQ    197            // NP*NP + 1
#define SEQP   224            // SEQ padded to multiple of 32 (K dim of AV GEMM)
#define EPSV   1e-5f

// LDS tile geometry: block tile 128(M) x 64(N), K-step 32 (bf16).
// Row pad = 40 halves (80 B = 20 dwords): 16-byte aligned rows for b128
// async stores, and the 16-row fragment-read sweep hits 16 distinct banks.
#define LDSPAD 40
#define A_ROWS 128
#define B_ROWS 64

typedef __attribute__((ext_vector_type(16))) __bf16 v16bf;
typedef __attribute__((ext_vector_type(8)))  float  v8f;

__device__ __forceinline__ unsigned short f2bf(float f) {
  // round-to-nearest-even fp32 -> bf16
  unsigned int u = __float_as_uint(f);
  u += 0x7FFFu + ((u >> 16) & 1u);
  return (unsigned short)(u >> 16);
}

__device__ __forceinline__ unsigned int pack2(float a, float b) {
  return (unsigned int)f2bf(a) | ((unsigned int)f2bf(b) << 16);
}

__device__ __forceinline__ float gelu_exact(float x) {
  return 0.5f * x * (1.0f + erff(x * 0.70710678118654752f));
}

// LDS byte offset from a generic pointer into __shared__ (ISA 10.2: the LDS
// aperture flat address carries the LDS offset in its low 32 bits).
__device__ __forceinline__ unsigned int lds_addr32(const void* p) {
  return (unsigned int)(unsigned long long)(uintptr_t)p;
}

// CDNA5 async global->LDS copies (ASYNCcnt-tracked; overlap with WMMA).
__device__ __forceinline__ void async_cp16(unsigned int lds_dst, const void* g) {
  asm volatile("global_load_async_to_lds_b128 %0, %1, off"
               :: "v"(lds_dst), "v"((unsigned long long)(uintptr_t)g)
               : "memory");
}
__device__ __forceinline__ void async_cp32(unsigned int lds_dst, const void* g) {
  asm volatile("global_load_async_to_lds_b128 %0, %1, off\n\t"
               "global_load_async_to_lds_b128 %0, %1, off offset:16"
               :: "v"(lds_dst), "v"((unsigned long long)(uintptr_t)g)
               : "memory");
}
__device__ __forceinline__ void wait_async0() {
  asm volatile("s_wait_asynccnt 0" ::: "memory");
}

// Stage one 128x32 A tile + 64x32 B tile (both bf16 in memory) into LDS via
// async b128 copies. Out-of-range rows are CLAMPED (not zeroed): C[i,j]
// depends only on A row i / W row j, and out-of-range i/j are never stored.
__device__ __forceinline__ void stage_async(
    unsigned short* __restrict__ sA, unsigned short* __restrict__ sB,
    const unsigned short* __restrict__ A, int lda,
    const unsigned short* __restrict__ W, int ldw,
    int m0, int n0, int k0, int M, int N, unsigned int tid)
{
  {
    const int row = (int)(tid >> 1);          // 0..127
    const int cb  = (int)((tid & 1u) << 4);   // 0 or 16 halves
    int gr = m0 + row;
    gr = (gr < M) ? gr : (M - 1);
    async_cp32(lds_addr32(sA + row * LDSPAD + cb),
               A + (long)gr * lda + k0 + cb);
  }
  {
    const int row = (int)(tid >> 2);          // 0..63
    const int cb  = (int)((tid & 3u) << 3);   // 0,8,16,24 halves
    int gn = n0 + row;
    gn = (gn < N) ? gn : (N - 1);
    async_cp16(lds_addr32(sB + row * LDSPAD + cb),
               W + (long)gn * ldw + k0 + cb);
  }
}

// ---------------------------------------------------------------------------
// Batched GEMM: C[z][M,N] = act( A[z][M,K] @ W[z][N,K]^T (+bias) ) (+resid)
//   A, W bf16 row-major (row strides lda/ldw); K % 32 == 0.
//   Batch z = blockIdx.z decomposed as (zo, zi) = (z/bdiv, z%bdiv);
//   per-matrix offsets = zo*xOut + zi*xIn (elements). C/resid share ldc+cOff.
//   Output: fp32 to C if Cbf==null, else bf16 to Cbf.
// Block tile 128x64, 256 threads = 8 waves (4x2); each wave owns 32x32 ->
// 4 x v_wmma_f32_16x16x32_bf16 per K-step; double-buffered async LDS staging.
// ---------------------------------------------------------------------------
__global__ __launch_bounds__(256)
void gemm_bf16_wmma(const unsigned short* __restrict__ A, int lda,
                    long aOut, long aIn,
                    const unsigned short* __restrict__ W, int ldw,
                    long wOut, long wIn,
                    const float* __restrict__ bias,
                    const float* __restrict__ resid,
                    float* __restrict__ C,
                    unsigned short* __restrict__ Cbf,
                    int ldc, long cOut, long cIn,
                    int M, int N, int K, int act, int bdiv)
{
  __shared__ __align__(16) unsigned short sA[2 * A_ROWS * LDSPAD];
  __shared__ __align__(16) unsigned short sB[2 * B_ROWS * LDSPAD];

  const int z  = (int)blockIdx.z;
  const int zo = z / bdiv, zi = z % bdiv;
  A += zo * aOut + zi * aIn;
  W += zo * wOut + zi * wIn;
  const long cbase = zo * cOut + zi * cIn;

  const unsigned int tid  = threadIdx.x;
  const unsigned int lane = tid & 31u;
  const unsigned int wid  = tid >> 5;      // 0..7
  const int wm = (int)(wid >> 1);          // 0..3  (M slab of 32)
  const int wn = (int)(wid & 1);           // 0..1  (N slab of 32)
  const int mf = (int)(lane & 15);         // row/col within fragment
  const int kb = (int)((lane >> 4) << 3);  // K base: 0 or 8 (lane group)
  const int r8 = kb;                       // C-row offset: 0 or 8 (same split)

  const int m0 = blockIdx.y * 128;
  const int n0 = blockIdx.x * 64;

  v8f acc00 = {}, acc01 = {}, acc10 = {}, acc11 = {};

  const int nk = K >> 5;
  stage_async(sA, sB, A, lda, W, ldw, m0, n0, 0, M, N, tid);
  wait_async0();
  __syncthreads();

  int p = 0;
  for (int kt = 0; kt < nk; ++kt) {
    if (kt + 1 < nk) {
      stage_async(sA + (1 - p) * A_ROWS * LDSPAD,
                  sB + (1 - p) * B_ROWS * LDSPAD,
                  A, lda, W, ldw, m0, n0, (kt + 1) << 5, M, N, tid);
    }

    const unsigned int* pAu =
        (const unsigned int*)(sA + p * A_ROWS * LDSPAD);
    const unsigned int* pBu =
        (const unsigned int*)(sB + p * B_ROWS * LDSPAD);

    union Frag { v16bf v; unsigned int u[8]; };
    Frag a0, a1, b0, b1;
#pragma unroll
    for (int j = 0; j < 8; ++j) {
      // K for VGPR j: (lane>=16 ? 8:0) + (j>=4 ? 16:0) + (j%4)*2
      const int kk = kb + ((j >> 2) << 4) + ((j & 3) << 1);
      const int kd = kk >> 1;                     // dword within 20-dword row
      a0.u[j] = pAu[(wm * 32 + mf)      * (LDSPAD / 2) + kd];
      a1.u[j] = pAu[(wm * 32 + 16 + mf) * (LDSPAD / 2) + kd];
      b0.u[j] = pBu[(wn * 32 + mf)      * (LDSPAD / 2) + kd];
      b1.u[j] = pBu[(wn * 32 + 16 + mf) * (LDSPAD / 2) + kd];
    }

    acc00 = __builtin_amdgcn_wmma_f32_16x16x32_bf16(
        false, a0.v, false, b0.v, (short)0, acc00, false, false);
    acc01 = __builtin_amdgcn_wmma_f32_16x16x32_bf16(
        false, a0.v, false, b1.v, (short)0, acc01, false, false);
    acc10 = __builtin_amdgcn_wmma_f32_16x16x32_bf16(
        false, a1.v, false, b0.v, (short)0, acc10, false, false);
    acc11 = __builtin_amdgcn_wmma_f32_16x16x32_bf16(
        false, a1.v, false, b1.v, (short)0, acc11, false, false);

    wait_async0();
    __syncthreads();
    p ^= 1;
  }

  // ---- epilogue: (+bias) (+gelu) (+residual), fp32 or bf16 store ----
#pragma unroll
  for (int mi = 0; mi < 2; ++mi) {
#pragma unroll
    for (int v = 0; v < 8; ++v) {
      const int row = m0 + wm * 32 + mi * 16 + r8 + v;
      if (row >= M) continue;
#pragma unroll
      for (int ni = 0; ni < 2; ++ni) {
        const int col = n0 + wn * 32 + ni * 16 + mf;
        if (col >= N) continue;
        float val;
        if (mi == 0 && ni == 0)      val = acc00[v];
        else if (mi == 0 && ni == 1) val = acc01[v];
        else if (mi == 1 && ni == 0) val = acc10[v];
        else                         val = acc11[v];
        const long idx = cbase + (long)row * ldc + col;
        if (bias)  val += bias[col];
        if (act)   val = gelu_exact(val);
        if (resid) val += resid[idx];
        if (Cbf) Cbf[idx] = f2bf(val);
        else     C[idx]   = val;
      }
    }
  }
}

// ---------------------------------------------------------------------------
// Weight fp32 -> bf16 convert (n % 4 == 0), 4 elems/thread
// ---------------------------------------------------------------------------
__global__ void cvt_bf16_kernel(const float* __restrict__ src,
                                unsigned short* __restrict__ dst, int n4)
{
  const int i = blockIdx.x * 256 + threadIdx.x;
  if (i >= n4) return;
  const float4 f = ((const float4*)src)[i];
  uint2 s;
  s.x = pack2(f.x, f.y);
  s.y = pack2(f.z, f.w);
  ((uint2*)dst)[i] = s;
}

// ---------------------------------------------------------------------------
// Patchify: images[B,C,224,224] -> bf16 patches[B,196,768]
// ---------------------------------------------------------------------------
__global__ void patchify_kernel(const float* __restrict__ img,
                                unsigned short* __restrict__ patches)
{
  const long idx = (long)blockIdx.x * 256 + threadIdx.x;
  const long total = (long)BVIT * NPAT * NPAT * DIM;
  if (idx >= total) return;
  const int f = (int)(idx % DIM);
  const long t = idx / DIM;
  const int p = (int)(t % (NPAT * NPAT));
  const int b = (int)(t / (NPAT * NPAT));
  const int py = p / NPAT, px = p % NPAT;
  const int c  = f >> 8;          // f / 256
  const int iy = (f >> 4) & 15;   // (f % 256) / 16
  const int ix = f & 15;
  patches[idx] = f2bf(
      img[(((long)b * CCH + c) * IMGSZ + (py * PSZ + iy)) * IMGSZ + (px * PSZ + ix)]);
}

// ---------------------------------------------------------------------------
// Assemble: out[b,s,:] = (s==0 ? cls : tokens[b,s-1,:]) + pe[s,:]   (fp32)
// ---------------------------------------------------------------------------
__global__ void assemble_kernel(const float* __restrict__ tokens,
                                const float* __restrict__ cls,
                                const float* __restrict__ pe,
                                float* __restrict__ out)
{
  const long idx = (long)blockIdx.x * 256 + threadIdx.x;
  const long total = (long)BVIT * SEQ * DIM;
  if (idx >= total) return;
  const int d = (int)(idx % DIM);
  const long t = idx / DIM;
  const int s = (int)(t % SEQ);
  const int b = (int)(t / SEQ);
  const float val = (s == 0) ? cls[d]
                             : tokens[((long)b * (SEQ - 1) + (s - 1)) * DIM + d];
  out[idx] = val + pe[(long)s * DIM + d];
}

// ---------------------------------------------------------------------------
// LayerNorm over last dim -> bf16 output (768 = 3*256 per thread)
// ---------------------------------------------------------------------------
__global__ __launch_bounds__(256)
void layernorm_kernel(const float* __restrict__ x, const float* __restrict__ g,
                      const float* __restrict__ b,
                      unsigned short* __restrict__ y)
{
  __shared__ float red[256];
  const int r = blockIdx.x;
  const unsigned int tid = threadIdx.x;
  const float* xr = x + (long)r * DIM;

  float vals[3];
  float lsum = 0.f;
#pragma unroll
  for (int i = 0; i < 3; ++i) { vals[i] = xr[tid + i * 256]; lsum += vals[i]; }
  red[tid] = lsum; __syncthreads();
  for (int off = 128; off > 0; off >>= 1) {
    if (tid < off) red[tid] += red[tid + off];
    __syncthreads();
  }
  const float mean = red[0] * (1.0f / DIM);
  __syncthreads();

  float lvar = 0.f;
#pragma unroll
  for (int i = 0; i < 3; ++i) { const float d = vals[i] - mean; lvar += d * d; }
  red[tid] = lvar; __syncthreads();
  for (int off = 128; off > 0; off >>= 1) {
    if (tid < off) red[tid] += red[tid + off];
    __syncthreads();
  }
  const float rstd = rsqrtf(red[0] * (1.0f / DIM) + EPSV);

  unsigned short* yr = y + (long)r * DIM;
#pragma unroll
  for (int i = 0; i < 3; ++i) {
    const int d = tid + i * 256;
    yr[d] = f2bf((vals[i] - mean) * rstd * g[d] + b[d]);
  }
}

// ---------------------------------------------------------------------------
// Attention softmax: one block per score row r in [0, 192*SEQ).
// Reads fp32 scores S[r][0..196], applies 1/sqrt(DH) scale, softmax,
// writes bf16 probabilities padded with zeros to SEQP columns.
// ---------------------------------------------------------------------------
__global__ __launch_bounds__(256)
void softmax_scale_kernel(const float* __restrict__ S,
                          unsigned short* __restrict__ P)
{
  __shared__ float red[256];
  const long r = blockIdx.x;
  const unsigned int tid = threadIdx.x;
  const float* sr = S + r * SEQ;
  unsigned short* pr = P + r * SEQP;
  const float scale = 0.125f;

  float lmax = -1e30f;
  for (int j = (int)tid; j < SEQ; j += 256) lmax = fmaxf(lmax, sr[j] * scale);
  red[tid] = lmax; __syncthreads();
  for (int off = 128; off > 0; off >>= 1) {
    if (tid < off) red[tid] = fmaxf(red[tid], red[tid + off]);
    __syncthreads();
  }
  const float mx = red[0]; __syncthreads();

  float lsum = 0.f;
  for (int j = (int)tid; j < SEQ; j += 256) lsum += __expf(sr[j] * scale - mx);
  red[tid] = lsum; __syncthreads();
  for (int off = 128; off > 0; off >>= 1) {
    if (tid < off) red[tid] += red[tid + off];
    __syncthreads();
  }
  const float inv = 1.0f / red[0];

  for (int j = (int)tid; j < SEQP; j += 256)
    pr[j] = (j < SEQ) ? f2bf(__expf(sr[j] * scale - mx) * inv)
                      : (unsigned short)0;
}

// ---------------------------------------------------------------------------
// Build V^T tiles: vT[z][e][j] = v[b, j, h*64+e] (bf16, zero-padded j>=SEQ)
// ---------------------------------------------------------------------------
__global__ void transpose_v_kernel(const unsigned short* __restrict__ vbf,
                                   unsigned short* __restrict__ vT)
{
  const long idx = (long)blockIdx.x * 256 + threadIdx.x;
  const long total = (long)BVIT * NHEAD * DHEAD * SEQP;
  if (idx >= total) return;
  const int j = (int)(idx % SEQP);
  const long t = idx / SEQP;
  const int e = (int)(t % DHEAD);
  const int z = (int)(t / DHEAD);
  const int b = z / NHEAD, h = z % NHEAD;
  vT[idx] = (j < SEQ)
      ? vbf[((long)b * SEQ + j) * DIM + h * DHEAD + e]
      : (unsigned short)0;
}

// ---------------------------------------------------------------------------
// Pool CLS rows to compact bf16 [B, D]
// ---------------------------------------------------------------------------
__global__ void pool_kernel(const float* __restrict__ out,
                            unsigned short* __restrict__ pooled)
{
  const int idx = blockIdx.x * 256 + threadIdx.x;
  if (idx >= BVIT * DIM) return;
  const int b = idx / DIM, d = idx % DIM;
  pooled[idx] = f2bf(out[(long)b * SEQ * DIM + d]);
}

// ---------------------------------------------------------------------------
// Final row softmax: logits[B,1000] -> probs
// ---------------------------------------------------------------------------
__global__ __launch_bounds__(256)
void softmax_kernel(const float* __restrict__ logits, float* __restrict__ out)
{
  __shared__ float red[256];
  const int b = blockIdx.x;
  const unsigned int tid = threadIdx.x;
  const float* lr = logits + (long)b * OUTC;

  float lmax = -1e30f;
  for (int j = (int)tid; j < OUTC; j += 256) lmax = fmaxf(lmax, lr[j]);
  red[tid] = lmax; __syncthreads();
  for (int off = 128; off > 0; off >>= 1) {
    if (tid < off) red[tid] = fmaxf(red[tid], red[tid + off]);
    __syncthreads();
  }
  const float mx = red[0]; __syncthreads();

  float lsum = 0.f;
  for (int j = (int)tid; j < OUTC; j += 256) lsum += __expf(lr[j] - mx);
  red[tid] = lsum; __syncthreads();
  for (int off = 128; off > 0; off >>= 1) {
    if (tid < off) red[tid] += red[tid + off];
    __syncthreads();
  }
  const float inv = 1.0f / red[0];
  for (int j = (int)tid; j < OUTC; j += 256)
    out[(long)b * OUTC + j] = __expf(lr[j] - mx) * inv;
}

// ---------------------------------------------------------------------------
extern "C" void kernel_launch(void* const* d_in, const int* in_sizes, int n_in,
                              void* d_out, int out_size, void* d_ws, size_t ws_size,
                              hipStream_t stream)
{
  (void)in_sizes; (void)n_in; (void)out_size; (void)ws_size;

  const float* images = (const float*)d_in[0];
  const float* Wm     = (const float*)d_in[1];
  const float* bm     = (const float*)d_in[2];
  const float* cls    = (const float*)d_in[3];
  const float* pe     = (const float*)d_in[4];
  const float* g1     = (const float*)d_in[5];
  const float* beta1  = (const float*)d_in[6];
  const float* Wq     = (const float*)d_in[7];
  const float* bq     = (const float*)d_in[8];
  const float* Wk     = (const float*)d_in[9];
  const float* bk     = (const float*)d_in[10];
  const float* Wv     = (const float*)d_in[11];
  const float* bv     = (const float*)d_in[12];
  const float* g2     = (const float*)d_in[13];
  const float* beta2  = (const float*)d_in[14];
  const float* W1     = (const float*)d_in[15];
  const float* b1     = (const float*)d_in[16];
  const float* W2     = (const float*)d_in[17];
  const float* b2     = (const float*)d_in[18];
  const float* Wc     = (const float*)d_in[19];
  const float* bc     = (const float*)d_in[20];

  const long NT  = (long)BVIT * SEQ;          // 3152 token rows
  const long NP2 = (long)BVIT * NPAT * NPAT;  // 3136 patch rows
  const int  NZ  = BVIT * NHEAD;              // 192 attention batches

  // workspace partition (~89 MB)
  char* wsb = (char*)d_ws;
  float*          outbuf = (float*)wsb;          wsb += NT * DIM * 4;          // residual fp32
  unsigned short* xbuf   = (unsigned short*)wsb; wsb += NT * DIM * 2;          // LN out bf16
  unsigned short* qbf    = (unsigned short*)wsb; wsb += NT * DIM * 2;
  unsigned short* kbf    = (unsigned short*)wsb; wsb += NT * DIM * 2;
  unsigned short* vbf    = (unsigned short*)wsb; wsb += NT * DIM * 2;
  unsigned short* vT     = (unsigned short*)wsb; wsb += (long)NZ * DHEAD * SEQP * 2;
  float*          Sbuf   = (float*)wsb;          wsb += (long)NZ * SEQ * SEQ * 4;
  unsigned short* hbuf   = (unsigned short*)wsb; wsb += NT * MLPD * 2;         // MLP hidden bf16
  unsigned short* wbuf   = (unsigned short*)wsb; wsb += (long)MLPD * DIM * 2;  // weight slab
  unsigned short* pooled = (unsigned short*)wsb; wsb += BVIT * DIM * 2;
  wsb = (char*)(((uintptr_t)wsb + 255) & ~(uintptr_t)255);
  float*          logits = (float*)wsb;          wsb += BVIT * OUTC * 4;
  unsigned short* patches = hbuf;                // alias (pre-loop only)
  float*          tokens  = Sbuf;                // alias (pre-loop only)
  unsigned short* Pbuf    = hbuf;                // alias (attention phase only)

  const dim3 blk(256);
  #define CVT(src, n) cvt_bf16_kernel<<<dim3((unsigned)(((n) / 4 + 255) / 256)), blk, 0, stream>>>((src), wbuf, (int)((n) / 4))
  // plain (unbatched) GEMM helper
  #define GEMM(A_, lda_, W_, bias_, resid_, C_, Cbf_, M_, N_, K_, act_)              \
    gemm_bf16_wmma<<<dim3(((N_) + 63) / 64, ((M_) + 127) / 128, 1), blk, 0, stream>>>( \
        (A_), (lda_), 0L, 0L, (W_), (K_), 0L, 0L, (bias_), (resid_), (C_), (Cbf_),   \
        (N_), 0L, 0L, (M_), (N_), (K_), (act_), 1)

  // 1) patchify -> bf16 patches
  {
    const long total = NP2 * DIM;
    patchify_kernel<<<dim3((unsigned)((total + 255) / 256)), blk, 0, stream>>>(
        images, patches);
  }
  // 2) patch embedding: tokens = patches @ Wm^T + bm   (M=3136, N=K=768)
  CVT(Wm, (long)DIM * DIM);
  GEMM(patches, DIM, wbuf, bm, nullptr, tokens, nullptr, (int)NP2, DIM, DIM, 0);
  // 3) out = concat(cls, tokens) + pe
  {
    const long total = NT * DIM;
    assemble_kernel<<<dim3((unsigned)((total + 255) / 256)), blk, 0, stream>>>(
        tokens, cls, pe, outbuf);
  }

  // 4) transformer layers
  for (int l = 0; l < LAYERS; ++l) {
    const float* g1l  = g1    + (long)l * DIM;
    const float* be1l = beta1 + (long)l * DIM;
    const float* g2l  = g2    + (long)l * DIM;
    const float* be2l = beta2 + (long)l * DIM;
    const float* Wql  = Wq + (long)l * DIM * DIM;  const float* bql = bq + (long)l * DIM;
    const float* Wkl  = Wk + (long)l * DIM * DIM;  const float* bkl = bk + (long)l * DIM;
    const float* Wvl  = Wv + (long)l * DIM * DIM;  const float* bvl = bv + (long)l * DIM;
    const float* W1l  = W1 + (long)l * MLPD * DIM; const float* b1l = b1 + (long)l * MLPD;
    const float* W2l  = W2 + (long)l * DIM * MLPD; const float* b2l = b2 + (long)l * DIM;

    layernorm_kernel<<<dim3((unsigned)NT), blk, 0, stream>>>(outbuf, g1l, be1l, xbuf);

    CVT(Wql, (long)DIM * DIM);
    GEMM(xbuf, DIM, wbuf, bql, nullptr, nullptr, qbf, (int)NT, DIM, DIM, 0);
    CVT(Wkl, (long)DIM * DIM);
    GEMM(xbuf, DIM, wbuf, bkl, nullptr, nullptr, kbf, (int)NT, DIM, DIM, 0);
    CVT(Wvl, (long)DIM * DIM);
    GEMM(xbuf, DIM, wbuf, bvl, nullptr, nullptr, vbf, (int)NT, DIM, DIM, 0);

    // ---- attention on the WMMA pipe ----
    // scores: S[z] = Q[z] @ K[z]^T   (M=N=SEQ, K=DHEAD)
    gemm_bf16_wmma<<<dim3((SEQ + 63) / 64, (SEQ + 127) / 128, NZ), blk, 0, stream>>>(
        qbf, DIM, (long)SEQ * DIM, (long)DHEAD,
        kbf, DIM, (long)SEQ * DIM, (long)DHEAD,
        nullptr, nullptr, Sbuf, nullptr,
        SEQ, (long)NHEAD * SEQ * SEQ, (long)SEQ * SEQ,
        SEQ, SEQ, DHEAD, 0, NHEAD);
    // P = softmax(S / 8), bf16, padded to SEQP columns
    softmax_scale_kernel<<<dim3((unsigned)(NZ * SEQ)), blk, 0, stream>>>(Sbuf, Pbuf);
    // V^T tiles (bf16, zero-padded)
    {
      const long total = (long)NZ * DHEAD * SEQP;
      transpose_v_kernel<<<dim3((unsigned)((total + 255) / 256)), blk, 0, stream>>>(
          vbf, vT);
    }
    // out[z-slice] += P[z] @ V[z]  (M=SEQ, N=DHEAD, K=SEQP; accumulate in place)
    gemm_bf16_wmma<<<dim3((DHEAD + 63) / 64, (SEQ + 127) / 128, NZ), blk, 0, stream>>>(
        Pbuf, SEQP, (long)NHEAD * SEQ * SEQP, (long)SEQ * SEQP,
        vT, SEQP, (long)NHEAD * DHEAD * SEQP, (long)DHEAD * SEQP,
        nullptr, outbuf, outbuf, nullptr,
        DIM, (long)SEQ * DIM, (long)DHEAD,
        SEQ, DHEAD, SEQP, 0, NHEAD);

    layernorm_kernel<<<dim3((unsigned)NT), blk, 0, stream>>>(outbuf, g2l, be2l, xbuf);

    CVT(W1l, (long)MLPD * DIM);
    GEMM(xbuf, DIM, wbuf, b1l, nullptr, nullptr, hbuf, (int)NT, MLPD, DIM, 1 /*gelu*/);
    CVT(W2l, (long)MLPD * DIM);
    GEMM(hbuf, MLPD, wbuf, b2l, outbuf, outbuf, nullptr, (int)NT, DIM, MLPD, 0);
  }

  // 5) classifier on CLS rows + softmax
  pool_kernel<<<dim3((BVIT * DIM + 255) / 256), blk, 0, stream>>>(outbuf, pooled);
  CVT(Wc, (long)OUTC * DIM);
  GEMM(pooled, DIM, wbuf, bc, nullptr, logits, nullptr, BVIT, OUTC, DIM, 0);
  softmax_kernel<<<dim3(BVIT), blk, 0, stream>>>(logits, (float*)d_out);
  #undef GEMM
  #undef CVT
}
